// MultiHeadCrossAttention_67963562492589
// MI455X (gfx1250) — matrix operand order repaired and checked by reading, not compiled, not run on hardware
//
#include <hip/hip_runtime.h>

// ---------------- WMMA types & helpers (gfx1250, wave32) ----------------
typedef __attribute__((ext_vector_type(16))) __bf16 v16bf;
typedef __attribute__((ext_vector_type(8)))  float  v8f;

struct alignas(16) U128 { unsigned int w[4]; };

union Frag16 {                 // 32 bytes: one 16-element bf16 WMMA operand
  v16bf v;
  unsigned short u[16];
  U128 q[2];
};

// Native f32 -> bf16 (RNE) via the hardware convert; avoids the 4-VALU
// software round-to-nearest-even emulation sequence.
__device__ __forceinline__ unsigned short f2bf(float f) {
  return __builtin_bit_cast(unsigned short, static_cast<__bf16>(f));
}

__device__ __forceinline__ unsigned int pack2bf(float lo, float hi) {
  return (unsigned int)f2bf(lo) | ((unsigned int)f2bf(hi) << 16);
}

__device__ __forceinline__ v8f wmma_bf16(v16bf a, v16bf b, v8f c) {
  // (neg_a, A, neg_b, B, c_mod, C, reuse_a, reuse_b)
  return __builtin_amdgcn_wmma_f32_16x16x32_bf16(false, a, false, b, (short)0, c,
                                                 false, false);
}

// ---------------- constants ----------------
#define BATCH 16
#define SEQ   1024
#define HEADS 4
#define DHEAD 256          // KD == VD == 256
#define NCOL  1024         // H*DHEAD, projection output width

// =====================================================================
// Kernel 0: zero the mean-accumulation buffer (graph-capture safe)
// =====================================================================
__global__ void zero_kernel(float* __restrict__ p, int n) {
  int i = blockIdx.x * blockDim.x + threadIdx.x;
  if (i < n) p[i] = 0.0f;
}

// =====================================================================
// Kernel 1: projection GEMM  C[M=16384, N=1024] = X[M,K] @ W[K,1024] + b
// Output bf16. storeVT==0 -> contiguous [B][2^20] (== [B][h][s2][d] view)
// storeVT==1 -> transposed V layout [B][h][d][s2] for contiguous attn B-loads
// block = 256 threads (8 waves); tile 128 rows x 64 cols; wave = 16x64.
// W k-slice staged TRANSPOSED in LDS so each B fragment is a contiguous
// 32-byte run -> 2x ds_load_b128 per fragment (no scalar LDS gathers).
// =====================================================================
__global__ __launch_bounds__(256) void proj_gemm_kernel(
    const float* __restrict__ X, const float* __restrict__ W,
    const float* __restrict__ bias, unsigned short* __restrict__ out,
    int K, int storeVT)
{
  __shared__ unsigned short ldsWT[64 * 32];  // [col][k] bf16, 4 KB

  const int tid  = threadIdx.x;
  const int wave = tid >> 5;
  const int lane = tid & 31;
  const int lh   = lane >> 4;    // half-wave select
  const int li   = lane & 15;

  const int mt = blockIdx.x >> 4;        // 0..127
  const int nt = blockIdx.x & 15;        // 0..15
  const int m0 = mt * 128 + wave * 16;   // this wave's 16 rows
  const int n0 = nt * 64;

  v8f acc[4] = {};

  // staging coords: thread handles k-rows {sr, sr+1} x cols {sc..sc+3}
  const int sr = (tid >> 4) << 1;        // 0,2,...,30
  const int sc = (tid & 15) << 2;        // 0,4,...,60

  const int kb16 = lh * 16;

  for (int dk = 0; dk < K; dk += 32) {
    { // stage W[dk..dk+32) x [n0..n0+64) -> LDS bf16, transposed [col][k]
      const float* wp0 = W + (size_t)(dk + sr) * NCOL + n0 + sc;
      const float* wp1 = wp0 + NCOL;
      float4 a0 = *(const float4*)wp0;
      float4 a1 = *(const float4*)wp1;
      unsigned int* lp = (unsigned int*)&ldsWT[0];
      const int base = ((sc * 32 + sr) >> 1);      // u32 index: col*16 + sr/2
      lp[base +  0] = pack2bf(a0.x, a1.x);
      lp[base + 16] = pack2bf(a0.y, a1.y);
      lp[base + 32] = pack2bf(a0.z, a1.z);
      lp[base + 48] = pack2bf(a0.w, a1.w);
      if (dk + 32 < K)                              // prefetch next k-slice
        __builtin_prefetch(wp0 + 32 * NCOL, 0, 3);
    }
    __syncthreads();

    // A fragment: row = m0+li; contiguous K: lane<16 K=0..15, lane>=16 K=16..31
    Frag16 a;
    {
      const float* ap = X + (size_t)(m0 + li) * K + dk + lh * 16;
      float4 f0 = ((const float4*)ap)[0];
      float4 f1 = ((const float4*)ap)[1];
      float4 f2 = ((const float4*)(ap + 8))[0];
      float4 f3 = ((const float4*)(ap + 8))[1];
      a.u[0] = f2bf(f0.x); a.u[1] = f2bf(f0.y); a.u[2]  = f2bf(f0.z); a.u[3]  = f2bf(f0.w);
      a.u[4] = f2bf(f1.x); a.u[5] = f2bf(f1.y); a.u[6]  = f2bf(f1.z); a.u[7]  = f2bf(f1.w);
      a.u[8] = f2bf(f2.x); a.u[9] = f2bf(f2.y); a.u[10] = f2bf(f2.z); a.u[11] = f2bf(f2.w);
      a.u[12] = f2bf(f3.x); a.u[13] = f2bf(f3.y); a.u[14] = f2bf(f3.z); a.u[15] = f2bf(f3.w);
    }

#pragma unroll
    for (int t = 0; t < 4; ++t) {
      Frag16 bf;                                   // contiguous run in ldsWT
      const int col = t * 16 + li;
      const unsigned short* bp = &ldsWT[col * 32 + kb16];
      bf.q[0] = ((const U128*)bp)[0];
      bf.q[1] = ((const U128*)bp)[1];
      acc[t] = wmma_bf16(a.v, bf.v, acc[t]);
    }
    __syncthreads();
  }

  // epilogue: +bias, convert bf16, store
#pragma unroll
  for (int t = 0; t < 4; ++t) {
    const int n = n0 + t * 16 + li;
    const float bb = bias[n];
#pragma unroll
    for (int r = 0; r < 8; ++r) {
      const int grow = m0 + lh * 8 + r;          // global M row = b*1024 + s
      const float val = acc[t][r] + bb;
      size_t idx;
      if (!storeVT) {
        idx = (size_t)grow * NCOL + n;           // contiguous [B][2^20]
      } else {
        const int b_ = grow >> 10;
        const int s  = grow & 1023;
        const int h_ = s >> 8;                   // torch-view head index
        const int s2 = ((s & 255) << 2) | (n >> 8);
        const int d2 = n & 255;
        idx = ((size_t)b_ << 20) + ((size_t)h_ << 18) + ((size_t)d2 << 10) + s2;
      }
      out[idx] = f2bf(val);
    }
  }
}

// =====================================================================
// Kernel 2: attention. One block per (b, h, 16-row q-tile).
// Phase 1: scores = tq @ tk^T  (wave w owns k-cols [w*128, w*128+128))
// Phase 2: row softmax (shuffle + LDS cross-wave reduce), write attn f32,
//          stage P as bf16 in LDS [16][1024]
// Phase 3: attended = P @ tv  (wave w owns d-cols [w*32, w*32+32)),
//          atomic f32 accumulation of the S-mean into meanbuf[b][1024]
// =====================================================================
__global__ __launch_bounds__(256) void attn_kernel(
    const unsigned short* __restrict__ Pq,
    const unsigned short* __restrict__ Pk,
    const unsigned short* __restrict__ PvT,
    float* __restrict__ attn_out,        // [B][H][S][S]
    float* __restrict__ meanbuf)         // [B][1024]
{
  __shared__ unsigned short ldsP[16 * 1024];   // 32 KB bf16 probabilities
  __shared__ float red[16][8];

  const int tid  = threadIdx.x;
  const int wave = tid >> 5;
  const int lane = tid & 31;
  const int lh = lane >> 4, li = lane & 15;
  const int kb16 = lh * 16;

  const int qt = blockIdx.x & 63;
  const int h  = (blockIdx.x >> 6) & 3;
  const int b  = blockIdx.x >> 8;
  const int q0 = qt * 16;

  const unsigned short* pq  = Pq  + ((size_t)b << 20) + ((size_t)h << 18);  // [s2][256]
  const unsigned short* pk  = Pk  + ((size_t)b << 20) + ((size_t)h << 18);  // [s2][256]
  const unsigned short* pvT = PvT + ((size_t)b << 20) + ((size_t)h << 18);  // [d][1024]

  // ---------- phase 1: scores ----------
  v8f acc[8] = {};
  for (int dk = 0; dk < 256; dk += 32) {
    Frag16 a;
    const unsigned short* ap = pq + (size_t)(q0 + li) * 256 + dk + lh * 16;
    a.q[0] = *(const U128*)ap;
    a.q[1] = *(const U128*)(ap + 8);
#pragma unroll
    for (int t = 0; t < 8; ++t) {
      Frag16 bf;                                   // B[d][k] = tk[k][d]: contiguous
      const int col = wave * 128 + t * 16 + li;
      const unsigned short* bp = pk + (size_t)col * 256 + dk + kb16;
      bf.q[0] = ((const U128*)bp)[0];
      bf.q[1] = ((const U128*)bp)[1];
      acc[t] = wmma_bf16(a.v, bf.v, acc[t]);
    }
  }

  // ---------- phase 2: softmax over 1024 k-cols ----------
  float gmax[8], inv[8];
#pragma unroll
  for (int r = 0; r < 8; ++r) {
    float v = -3.0e38f;
#pragma unroll
    for (int t = 0; t < 8; ++t) v = fmaxf(v, acc[t][r]);
#pragma unroll
    for (int m = 1; m < 16; m <<= 1) v = fmaxf(v, __shfl_xor(v, m, 32));
    if (li == 0) red[lh * 8 + r][wave] = v;
  }
  __syncthreads();
#pragma unroll
  for (int r = 0; r < 8; ++r) {
    const int row = lh * 8 + r;
    float v = red[row][0];
#pragma unroll
    for (int w2 = 1; w2 < 8; ++w2) v = fmaxf(v, red[row][w2]);
    gmax[r] = v;
  }
  __syncthreads();

  float ps[8] = {};
#pragma unroll
  for (int t = 0; t < 8; ++t)
#pragma unroll
    for (int r = 0; r < 8; ++r) {
      const float e = __expf(acc[t][r] - gmax[r]);
      acc[t][r] = e;
      ps[r] += e;
    }
#pragma unroll
  for (int r = 0; r < 8; ++r) {
    float v = ps[r];
#pragma unroll
    for (int m = 1; m < 16; m <<= 1) v += __shfl_xor(v, m, 32);
    if (li == 0) red[lh * 8 + r][wave] = v;
  }
  __syncthreads();
#pragma unroll
  for (int r = 0; r < 8; ++r) {
    const int row = lh * 8 + r;
    float v = 0.0f;
#pragma unroll
    for (int w2 = 0; w2 < 8; ++w2) v += red[row][w2];
    inv[r] = __builtin_amdgcn_rcpf(v);
  }

  // write attn probs (f32, part of the output) + stage bf16 copy in LDS
  float* aout = attn_out + (((size_t)(b * 4 + h) * 1024 + q0) << 10);
#pragma unroll
  for (int t = 0; t < 8; ++t) {
    const int col = wave * 128 + t * 16 + li;
#pragma unroll
    for (int r = 0; r < 8; ++r) {
      const int row = lh * 8 + r;
      const float p = acc[t][r] * inv[r];
      aout[((size_t)row << 10) + col] = p;
      ldsP[row * 1024 + col] = f2bf(p);
    }
  }
  __syncthreads();

  // ---------- phase 3: attended = P @ tv ----------
  v8f oacc[2] = {};
  for (int k0 = 0; k0 < 1024; k0 += 32) {
    Frag16 a;
    const unsigned short* lp = &ldsP[li * 1024 + k0 + lh * 16];
    a.q[0] = *(const U128*)lp;
    a.q[1] = *(const U128*)(lp + 8);
#pragma unroll
    for (int t = 0; t < 2; ++t) {
      Frag16 bf;                                 // B[k][d] = tvT[d][k]: contiguous
      const int col = wave * 32 + t * 16 + li;
      const unsigned short* bp = pvT + ((size_t)col << 10) + k0 + kb16;
      bf.q[0] = ((const U128*)bp)[0];
      bf.q[1] = ((const U128*)bp)[1];
      oacc[t] = wmma_bf16(a.v, bf.v, oacc[t]);
    }
  }

  // mean over the [B,S',1024] view: target col c' = (s2&3)*256 + d, s2&3 == r&3
  const float scale = 1.0f / 1024.0f;
#pragma unroll
  for (int t = 0; t < 2; ++t) {
    float part[4] = {0.f, 0.f, 0.f, 0.f};
#pragma unroll
    for (int r = 0; r < 8; ++r) part[r & 3] += oacc[t][r];
    const int d = wave * 32 + t * 16 + li;
#pragma unroll
    for (int j = 0; j < 4; ++j)
      atomicAdd(&meanbuf[b * 1024 + j * 256 + d], part[j] * scale);
  }
}

// =====================================================================
// Kernel 3: head.  out[b] = (mean[b] @ Wo + bo) @ W1 + b1    (tiny)
// =====================================================================
__global__ __launch_bounds__(256) void head_kernel(
    const float* __restrict__ meanbuf, const float* __restrict__ Wo,
    const float* __restrict__ bo, const float* __restrict__ W1,
    const float* __restrict__ b1, float* __restrict__ out)
{
  __shared__ float sm[1024];
  __shared__ float st[2048];
  const int b = blockIdx.x, tid = threadIdx.x;
  for (int i = tid; i < 1024; i += 256) sm[i] = meanbuf[b * 1024 + i];
  __syncthreads();
  for (int c = tid; c < 2048; c += 256) {
    float a = bo[c];
    for (int j = 0; j < 1024; ++j) a = fmaf(sm[j], Wo[(size_t)j * 2048 + c], a);
    st[c] = a;
  }
  __syncthreads();
  for (int c = tid; c < 128; c += 256) {
    float a = b1[c];
    for (int j = 0; j < 2048; ++j) a = fmaf(st[j], W1[(size_t)j * 128 + c], a);
    out[b * 128 + c] = a;
  }
}

// =====================================================================
extern "C" void kernel_launch(void* const* d_in, const int* in_sizes, int n_in,
                              void* d_out, int out_size, void* d_ws, size_t ws_size,
                              hipStream_t stream) {
  (void)in_sizes; (void)n_in; (void)out_size; (void)ws_size;
  const float* query = (const float*)d_in[0];
  const float* key   = (const float*)d_in[1];
  const float* value = (const float*)d_in[2];
  const float* Wq = (const float*)d_in[3];  const float* bq = (const float*)d_in[4];
  const float* Wk = (const float*)d_in[5];  const float* bk = (const float*)d_in[6];
  const float* Wv = (const float*)d_in[7];  const float* bv = (const float*)d_in[8];
  const float* Wo = (const float*)d_in[9];  const float* bo = (const float*)d_in[10];
  const float* W1 = (const float*)d_in[11]; const float* b1 = (const float*)d_in[12];
  // d_in[13] = random_len (unused, == S)

  // workspace layout: meanbuf 64KB | Pq 32MB | Pk 32MB | PvT 32MB
  char* ws = (char*)d_ws;
  float* meanbuf = (float*)ws;
  unsigned short* Pq  = (unsigned short*)(ws + (64 << 10));
  unsigned short* Pk  = Pq + ((size_t)16 << 20);
  unsigned short* PvT = Pk + ((size_t)16 << 20);

  float* out = (float*)d_out;                 // [16,128]
  float* attn_out = out + BATCH * 128;        // [16,4,1024,1024]

  zero_kernel<<<64, 256, 0, stream>>>(meanbuf, BATCH * 1024);

  proj_gemm_kernel<<<2048, 256, 0, stream>>>(query, Wq, bq, Pq, 128, 0);
  proj_gemm_kernel<<<2048, 256, 0, stream>>>(key,   Wk, bk, Pk, 256, 0);
  proj_gemm_kernel<<<2048, 256, 0, stream>>>(value, Wv, bv, PvT, 256, 1);

  attn_kernel<<<4096, 256, 0, stream>>>(Pq, Pk, PvT, attn_out, meanbuf);

  head_kernel<<<16, 256, 0, stream>>>(meanbuf, Wo, bo, W1, b1, out);
}
